// ClusterMambaLayer_63917703299107
// MI455X (gfx1250) — matrix operand order repaired and checked by reading, not compiled, not run on hardware
//
#include <hip/hip_runtime.h>
#include <math.h>

// ---- problem constants ----
#define D_MODEL 128
#define NCLU    8
#define DIN     256
#define DST     16
#define DTR     8
#define BATCH   2
#define NPIX    4096
#define NPAIR   (NCLU*BATCH)        // 16 (k,b) pairs
#define ROWS    (NPAIR*NPIX)        // 65536 rows through the cluster-mamba
#define DBC_LD  48                  // 40 padded to 48 for 16-wide WMMA tiles
#define CHUNK   64                  // scan chunk length
#define NCHUNK  (NPIX/CHUNK)        // 64 chunks per (k,b)

typedef __attribute__((ext_vector_type(16))) __bf16       v16bf;
typedef __attribute__((ext_vector_type(8)))  float        v8f;
typedef __attribute__((ext_vector_type(8)))  unsigned int v8u;

__device__ __forceinline__ unsigned short f2bf(float f) {
  unsigned u = __float_as_uint(f);
  u += 0x7FFFu + ((u >> 16) & 1u);          // round-to-nearest-even
  return (unsigned short)(u >> 16);
}
__device__ __forceinline__ float bf2f(unsigned short h) {
  return __uint_as_float(((unsigned)h) << 16);
}
__device__ __forceinline__ float sigmoidf_(float x){ return 1.f/(1.f+__expf(-x)); }
__device__ __forceinline__ float siluf_(float x){ return x*sigmoidf_(x); }
__device__ __forceinline__ float softplusf_(float x){ return x>20.f ? x : log1pf(__expf(x)); }
__device__ __forceinline__ float geluf_(float x){ return 0.5f*x*(1.f+erff(x*0.70710678118654752f)); }
__device__ __forceinline__ float wsum(float v){
  #pragma unroll
  for (int m=16;m>=1;m>>=1) v += __shfl_xor(v,m,32);
  return v;
}

// ======================================================================
// bf16 WMMA GEMM: C[M,N](f32) = A[M,K](bf16) @ W[N,K](bf16)^T
// one wave32 per 16x16 tile; K stepped by 32 via v_wmma_f32_16x16x32_bf16
// ======================================================================
__global__ void wmma_gemm_bf16(const unsigned short* __restrict__ A,
                               const unsigned short* __restrict__ W,
                               float* __restrict__ C,
                               int M, int N, int K)
{
  const int lane = threadIdx.x & 31;
  const int tile = blockIdx.x;
  const int ntn  = N >> 4;
  const int tm   = (tile / ntn) << 4;
  const int tn   = (tile % ntn) << 4;
  const int half = lane >> 4;     // 0: lanes 0-15, 1: lanes 16-31
  const int l15  = lane & 15;

  v8f acc = {};
  for (int k0 = 0; k0 < K; k0 += 32) {
    // A (16x32 bf16, MxK): lane holds row tm+l15.
    // VGPR p<4 : K = k0 + 8*half + {2p, 2p+1}
    // VGPR p>=4: K = k0 + 8*half + 16 + {2(p-4), 2(p-4)+1}
    const unsigned short* arow = A + (size_t)(tm + l15) * K + k0 + (half << 3);
    v8u au, bu;
    #pragma unroll
    for (int p = 0; p < 4; ++p) {
      au[p]     = *(const unsigned int*)(arow + 2*p);
      au[4 + p] = *(const unsigned int*)(arow + 16 + 2*p);
    }
    // B (32x16 bf16, KxN): lane holds column tn+l15; same K mapping as A.
    const unsigned short* wrow = W + (size_t)(tn + l15) * K + k0 + (half << 3);
    #pragma unroll
    for (int i = 0; i < 4; ++i) {
      bu[i]     = *(const unsigned int*)(wrow + 2*i);
      bu[4 + i] = *(const unsigned int*)(wrow + 16 + 2*i);
    }

    v16bf a = __builtin_bit_cast(v16bf, au);
    v16bf b = __builtin_bit_cast(v16bf, bu);
    acc = __builtin_amdgcn_wmma_f32_16x16x32_bf16(false, a, false, b,
                                                  (short)0, acc, false, false);
  }
  // C/D layout: VGPR v holds (M = tm + v + 8*half, N = tn + l15)
  #pragma unroll
  for (int v = 0; v < 8; ++v)
    C[(size_t)(tm + v + (half << 3)) * N + tn + l15] = acc[v];
}

// ======================================================================
// weight conversions
// ======================================================================
__global__ void k_cvt_bf16(const float* __restrict__ s, unsigned short* __restrict__ d, int n) {
  int e = blockIdx.x * blockDim.x + threadIdx.x;
  if (e < n) d[e] = f2bf(s[e]);
}
__global__ void k_pad_wx(const float* __restrict__ wx, unsigned short* __restrict__ out) {
  int e = blockIdx.x * blockDim.x + threadIdx.x;   // 48*256
  int row = e >> 8, col = e & 255;
  out[e] = (row < 40) ? f2bf(wx[row * 256 + col]) : (unsigned short)0;
}

// ======================================================================
// hard gumbel assignment + masked bf16 inputs   (wave per pixel)
// ======================================================================
__global__ void k_assign(const float* __restrict__ x, const float* __restrict__ gum,
                         const float* __restrict__ cen,
                         int* __restrict__ idx, unsigned short* __restrict__ masked)
{
  int wave = (blockIdx.x * blockDim.x + threadIdx.x) >> 5;   // pixel id = b*4096+n
  int lane = threadIdx.x & 31;
  int b = wave >> 12, n = wave & 4095;
  const float* xr = x + (size_t)wave * 128;
  float xv[4];
  #pragma unroll
  for (int j = 0; j < 4; ++j) xv[j] = xr[lane * 4 + j];
  float best = -1e30f; int bk = 0;
  #pragma unroll
  for (int k = 0; k < 8; ++k) {
    const float* cr = cen + k * 128;
    float p = 0.f;
    #pragma unroll
    for (int j = 0; j < 4; ++j) { float d = xv[j] - cr[lane * 4 + j]; p += d * d; }
    p = wsum(p);
    float logit = -sqrtf(fmaxf(p, 0.f)) + gum[(size_t)wave * 8 + k];
    if (logit > best) { best = logit; bk = k; }
  }
  if (lane == 0) idx[wave] = bk;
  #pragma unroll
  for (int k = 0; k < 8; ++k) {
    size_t r = ((size_t)(k * 2 + b) * NPIX + n) * 128;
    #pragma unroll
    for (int j = 0; j < 4; ++j)
      masked[r + lane * 4 + j] = (k == bk) ? f2bf(xv[j]) : (unsigned short)0;
  }
}

// ======================================================================
// depthwise causal conv(4) + bias + silu  -> bf16 xi
// ======================================================================
__global__ void k_conv_silu(const float* __restrict__ xz, const float* __restrict__ cw,
                            const float* __restrict__ cb, unsigned short* __restrict__ xi_bf)
{
  size_t e = (size_t)blockIdx.x * blockDim.x + threadIdx.x;   // ROWS*256
  int c = (int)(e & 255); size_t r = e >> 8; int l = (int)(r & 4095);
  float acc = cb[c];
  #pragma unroll
  for (int j = 0; j < 4; ++j) {
    int ll = l - 3 + j;
    if (ll >= 0) acc += cw[c * 4 + j] * xz[(r + (size_t)(ll - l)) * 512 + c];
  }
  xi_bf[e] = f2bf(siluf_(acc));
}

// ======================================================================
// dt = softplus(dbc[:, :8] @ wdt^T + bdt)
// ======================================================================
__global__ void k_dt(const float* __restrict__ dbc, const float* __restrict__ wdt,
                     const float* __restrict__ bdt, float* __restrict__ dt)
{
  size_t e = (size_t)blockIdx.x * blockDim.x + threadIdx.x;   // ROWS*256
  int d = (int)(e & 255); size_t r = e >> 8;
  const float* q = dbc + r * DBC_LD;
  float acc = bdt[d];
  #pragma unroll
  for (int j = 0; j < 8; ++j) acc += q[j] * wdt[d * 8 + j];
  dt[e] = softplusf_(acc);
}

// ======================================================================
// chunked (Blelloch-style) selective scan.
// Recurrence per (pair p, channel d, state s):
//   h_l = e_l*h_{l-1} + dt_l*x_l*B_l[s],  e_l = exp(dt_l*a[d][s])
// Pass 1: per chunk, compute E = prod(e_l) and h_end with h_in=0.
// Pass 2: sequential combine over the 64 chunk summaries -> h_in per chunk.
// Pass 3: replay each chunk from h_in, emit y (in-place over dt buffer).
// B/C staged once per chunk in LDS -> zero barriers in the hot loops.
// ======================================================================
__global__ void k_scan1(const float* __restrict__ dbc, const float* __restrict__ dt,
                        const unsigned short* __restrict__ xi_bf,
                        const float* __restrict__ alog,
                        float* __restrict__ Eb, float* __restrict__ H0b)
{
  const int blk = blockIdx.x;          // p*NCHUNK + c
  const int d   = threadIdx.x;
  const size_t r0 = (size_t)(blk >> 6) * NPIX + (size_t)(blk & 63) * CHUNK;

  // prefetch this chunk's dt / xi rows (global_prefetch_b8)
  __builtin_prefetch(dt   + (r0 + (d >> 2)) * 256 + (d & 3) * 64, 0, 3);
  __builtin_prefetch(xi_bf + (r0 + (d >> 2)) * 256 + (d & 3) * 64, 0, 3);

  __shared__ float sB[CHUNK][16];
  for (int e = d; e < CHUNK * 16; e += 256) {
    int l = e >> 4, s = e & 15;
    sB[l][s] = dbc[(r0 + l) * DBC_LD + 8 + s];
  }
  __syncthreads();

  float a[16], h[16], E[16];
  #pragma unroll
  for (int s = 0; s < 16; ++s) { a[s] = -__expf(alog[d * 16 + s]); h[s] = 0.f; E[s] = 1.f; }
  for (int l = 0; l < CHUNK; ++l) {
    size_t r = r0 + l;
    float dtv = dt[r * 256 + d];
    float x   = bf2f(xi_bf[r * 256 + d]);
    float bx  = dtv * x;
    #pragma unroll
    for (int s = 0; s < 16; ++s) {
      float el = __expf(dtv * a[s]);
      h[s] = el * h[s] + bx * sB[l][s];
      E[s] *= el;
    }
  }
  size_t base = ((size_t)blk * 256 + d) * 16;
  #pragma unroll
  for (int s = 0; s < 16; ++s) { Eb[base + s] = E[s]; H0b[base + s] = h[s]; }
}

__global__ void k_scan2(const float* __restrict__ Eb, const float* __restrict__ H0b,
                        float* __restrict__ HIN)
{
  const int p = blockIdx.x, d = threadIdx.x;
  float h[16];
  #pragma unroll
  for (int s = 0; s < 16; ++s) h[s] = 0.f;
  for (int c = 0; c < NCHUNK; ++c) {
    size_t base = (((size_t)p * NCHUNK + c) * 256 + d) * 16;
    #pragma unroll
    for (int s = 0; s < 16; ++s) HIN[base + s] = h[s];
    #pragma unroll
    for (int s = 0; s < 16; ++s) h[s] = Eb[base + s] * h[s] + H0b[base + s];
  }
}

__global__ void k_scan3(const float* __restrict__ dbc, float* __restrict__ dt_ys,
                        const unsigned short* __restrict__ xi_bf,
                        const float* __restrict__ alog, const float* __restrict__ dpar,
                        const float* __restrict__ HIN)
{
  const int blk = blockIdx.x;
  const int d   = threadIdx.x;
  const size_t r0 = (size_t)(blk >> 6) * NPIX + (size_t)(blk & 63) * CHUNK;

  __builtin_prefetch(dt_ys + (r0 + (d >> 2)) * 256 + (d & 3) * 64, 0, 3);
  __builtin_prefetch(xi_bf + (r0 + (d >> 2)) * 256 + (d & 3) * 64, 0, 3);

  __shared__ float sB[CHUNK][16], sC[CHUNK][16];
  for (int e = d; e < CHUNK * 16; e += 256) {
    int l = e >> 4, s = e & 15;
    sB[l][s] = dbc[(r0 + l) * DBC_LD + 8 + s];
    sC[l][s] = dbc[(r0 + l) * DBC_LD + 24 + s];
  }
  __syncthreads();

  float a[16], h[16];
  size_t base = ((size_t)blk * 256 + d) * 16;
  #pragma unroll
  for (int s = 0; s < 16; ++s) { a[s] = -__expf(alog[d * 16 + s]); h[s] = HIN[base + s]; }
  float dp = dpar[d];
  for (int l = 0; l < CHUNK; ++l) {
    size_t r = r0 + l;
    float dtv = dt_ys[r * 256 + d];
    float x   = bf2f(xi_bf[r * 256 + d]);
    float y = dp * x, bx = dtv * x;
    #pragma unroll
    for (int s = 0; s < 16; ++s) {
      h[s] = __expf(dtv * a[s]) * h[s] + bx * sB[l][s];
      y += h[s] * sC[l][s];
    }
    dt_ys[r * 256 + d] = y;
  }
}

// ======================================================================
// gate: yg = bf16( y * silu(z) )   (reuses xi_bf buffer)
// ======================================================================
__global__ void k_gate(const float* __restrict__ ys, const float* __restrict__ xz,
                       unsigned short* __restrict__ yg)
{
  size_t e = (size_t)blockIdx.x * blockDim.x + threadIdx.x;   // ROWS*256
  int d = (int)(e & 255); size_t r = e >> 8;
  float z = xz[r * 512 + 256 + d];
  yg[e] = f2bf(ys[e] * siluf_(z));
}

// ======================================================================
// in-place layernorm over last dim (128), wave per row
// ======================================================================
__global__ void k_layernorm(float* __restrict__ u, const float* __restrict__ g,
                            const float* __restrict__ bb)
{
  int wave = (blockIdx.x * blockDim.x + threadIdx.x) >> 5;
  int lane = threadIdx.x & 31;
  float* r = u + (size_t)wave * 128;
  float v[4]; float s = 0.f;
  #pragma unroll
  for (int j = 0; j < 4; ++j) { v[j] = r[lane * 4 + j]; s += v[j]; }
  s = wsum(s);
  float mean = s * (1.f / 128.f);
  float q = 0.f;
  #pragma unroll
  for (int j = 0; j < 4; ++j) { float dd = v[j] - mean; q += dd * dd; }
  q = wsum(q);
  float inv = rsqrtf(q * (1.f / 128.f) + 1e-5f);
  #pragma unroll
  for (int j = 0; j < 4; ++j) { int d = lane * 4 + j; r[d] = (v[j] - mean) * inv * g[d] + bb[d]; }
}

// ======================================================================
// attention-score MLP (128->64 gelu ->1) + hard mask penalty, wave per row
// ======================================================================
__global__ void k_score(const float* __restrict__ upd, const float* __restrict__ w1,
                        const float* __restrict__ b1, const float* __restrict__ w2,
                        const float* __restrict__ b2, const int* __restrict__ idx,
                        float* __restrict__ scores)
{
  int wave = (blockIdx.x * blockDim.x + threadIdx.x) >> 5;
  int lane = threadIdx.x & 31;
  const float* r = upd + (size_t)wave * 128;
  float part = 0.f;
  #pragma unroll
  for (int jj = 0; jj < 2; ++jj) {
    int j = lane + jj * 32;
    float acc = b1[j];
    for (int d = 0; d < 128; ++d) acc += r[d] * w1[j * 128 + d];
    part += geluf_(acc) * w2[j];
  }
  part = wsum(part);
  if (lane == 0) {
    int k = wave >> 13;
    int bn = wave & 8191;            // b*4096+n
    float pen = (idx[bn] == k) ? 0.f : 1e9f;
    scores[wave] = part + b2[0] - pen;
  }
}

// ======================================================================
// softmax over n=4096 per (k,b)
// ======================================================================
__global__ void k_softmax_aw(const float* __restrict__ scores, float* __restrict__ aw)
{
  int p = blockIdx.x, t = threadIdx.x;
  __shared__ float red[256];
  const float* s = scores + (size_t)p * NPIX;
  float mx = -1e38f;
  for (int n = t; n < NPIX; n += 256) mx = fmaxf(mx, s[n]);
  red[t] = mx; __syncthreads();
  for (int o = 128; o > 0; o >>= 1) { if (t < o) red[t] = fmaxf(red[t], red[t + o]); __syncthreads(); }
  mx = red[0]; __syncthreads();
  float sum = 0.f;
  for (int n = t; n < NPIX; n += 256) sum += __expf(s[n] - mx);
  red[t] = sum; __syncthreads();
  for (int o = 128; o > 0; o >>= 1) { if (t < o) red[t] += red[t + o]; __syncthreads(); }
  float inv = 1.f / red[0];
  for (int n = t; n < NPIX; n += 256) aw[(size_t)p * NPIX + n] = __expf(s[n] - mx) * inv;
}

// ======================================================================
// reps[b][k][d] = sum_n aw * upd   (block per (k,b), 128 threads)
// ======================================================================
__global__ void k_reps(const float* __restrict__ aw, const float* __restrict__ upd,
                       float* __restrict__ reps)
{
  int p = blockIdx.x, d = threadIdx.x;
  int k = p >> 1, b = p & 1;
  const float* a = aw + (size_t)p * NPIX;
  const float* u = upd + (size_t)p * NPIX * 128;
  float acc = 0.f;
  for (int n = 0; n < NPIX; ++n) acc += a[n] * u[(size_t)n * 128 + d];
  reps[(size_t)(b * 8 + k) * 128 + d] = acc;
}

// ======================================================================
// global Mamba over the 8 cluster reps (L=8) + gn layernorm + fg MLP.
// one block per batch, everything staged in LDS (tiny sizes).
// ======================================================================
__global__ void k_global(const float* __restrict__ reps,
                         const float* __restrict__ win, const float* __restrict__ cw,
                         const float* __restrict__ cb,  const float* __restrict__ wx,
                         const float* __restrict__ wdt, const float* __restrict__ bdt,
                         const float* __restrict__ alog,const float* __restrict__ dpar,
                         const float* __restrict__ wout,
                         const float* __restrict__ gng, const float* __restrict__ gnb,
                         const float* __restrict__ fw1, const float* __restrict__ fb1,
                         const float* __restrict__ fw2, const float* __restrict__ fb2,
                         float* __restrict__ ctx_out, float* __restrict__ fwt_out)
{
  const int b = blockIdx.x, t = threadIdx.x;
  __shared__ float s_xz[8 * 512];
  __shared__ float s_xi[8 * 256];
  __shared__ float s_dbc[8 * 40];
  __shared__ float s_dt[8 * 256];
  __shared__ float s_y[8 * 256];
  __shared__ float s_ctx[8 * 128];
  __shared__ float s_stat[16];

  for (int e = t; e < 8 * 512; e += 256) {
    int l = e >> 9, c = e & 511;
    const float* xr = reps + (size_t)(b * 8 + l) * 128;
    float acc = 0.f;
    for (int d = 0; d < 128; ++d) acc += xr[d] * win[c * 128 + d];
    s_xz[e] = acc;
  }
  __syncthreads();
  for (int e = t; e < 8 * 256; e += 256) {
    int l = e >> 8, c = e & 255;
    float acc = cb[c];
    #pragma unroll
    for (int j = 0; j < 4; ++j) { int ll = l - 3 + j; if (ll >= 0) acc += cw[c * 4 + j] * s_xz[ll * 512 + c]; }
    s_xi[e] = siluf_(acc);
  }
  __syncthreads();
  for (int e = t; e < 8 * 40; e += 256) {
    int l = e / 40, j = e % 40;
    float acc = 0.f;
    for (int d = 0; d < 256; ++d) acc += s_xi[l * 256 + d] * wx[j * 256 + d];
    s_dbc[e] = acc;
  }
  __syncthreads();
  for (int e = t; e < 8 * 256; e += 256) {
    int l = e >> 8, d = e & 255;
    float acc = bdt[d];
    #pragma unroll
    for (int j = 0; j < 8; ++j) acc += s_dbc[l * 40 + j] * wdt[d * 8 + j];
    s_dt[e] = softplusf_(acc);
  }
  __syncthreads();
  { // scan + gate, channel per thread
    int d = t; float h[16], a[16];
    #pragma unroll
    for (int s = 0; s < 16; ++s) { h[s] = 0.f; a[s] = -__expf(alog[d * 16 + s]); }
    float dp = dpar[d];
    for (int l = 0; l < 8; ++l) {
      float dtv = s_dt[l * 256 + d], x = s_xi[l * 256 + d];
      float y = dp * x, bx = dtv * x;
      #pragma unroll
      for (int s = 0; s < 16; ++s) {
        h[s] = __expf(dtv * a[s]) * h[s] + bx * s_dbc[l * 40 + 8 + s];
        y += h[s] * s_dbc[l * 40 + 24 + s];
      }
      float z = s_xz[l * 512 + 256 + d];
      s_y[l * 256 + d] = y * siluf_(z);
    }
  }
  __syncthreads();
  for (int e = t; e < 8 * 128; e += 256) {
    int l = e >> 7, dm = e & 127;
    float acc = 0.f;
    for (int d = 0; d < 256; ++d) acc += s_y[l * 256 + d] * wout[dm * 256 + d];
    s_ctx[e] = acc;
  }
  __syncthreads();
  if (t < 8) {
    float m = 0.f;
    for (int dm = 0; dm < 128; ++dm) m += s_ctx[t * 128 + dm];
    m *= (1.f / 128.f);
    float v = 0.f;
    for (int dm = 0; dm < 128; ++dm) { float dd = s_ctx[t * 128 + dm] - m; v += dd * dd; }
    s_stat[t] = m; s_stat[8 + t] = rsqrtf(v * (1.f / 128.f) + 1e-5f);
  }
  __syncthreads();
  for (int e = t; e < 8 * 128; e += 256) {
    int l = e >> 7, dm = e & 127;
    float val = (s_ctx[e] - s_stat[l]) * s_stat[8 + l] * gng[dm] + gnb[dm];
    s_ctx[e] = val;
    ctx_out[(size_t)(b * 8 + l) * 128 + dm] = val;
  }
  __syncthreads();
  if (t < 8) {
    int l = t; float acc2 = fb2[0];
    for (int j = 0; j < 64; ++j) {
      float acc = fb1[j];
      for (int dm = 0; dm < 128; ++dm) acc += s_ctx[l * 128 + dm] * fw1[j * 128 + dm];
      acc2 += geluf_(acc) * fw2[j];
    }
    fwt_out[b * 8 + l] = sigmoidf_(acc2);
  }
}

// ======================================================================
// final combine (hard masks): out = upd[k*] + fwt[b,k*]*ctx[b,k*]
// ======================================================================
__global__ void k_out(const float* __restrict__ upd, const int* __restrict__ idx,
                      const float* __restrict__ fwt, const float* __restrict__ ctx,
                      float* __restrict__ out)
{
  size_t e = (size_t)blockIdx.x * blockDim.x + threadIdx.x;  // 2*4096*128
  int d = (int)(e & 127); size_t bn = e >> 7;
  int b = (int)(bn >> 12), n = (int)(bn & 4095);
  int k = idx[bn];
  size_t r = (size_t)(k * 2 + b) * NPIX + n;
  out[e] = upd[r * 128 + d] + fwt[b * 8 + k] * ctx[(size_t)(b * 8 + k) * 128 + d];
}

// ======================================================================
extern "C" void kernel_launch(void* const* d_in, const int* in_sizes, int n_in,
                              void* d_out, int out_size, void* d_ws, size_t ws_size,
                              hipStream_t stream)
{
  (void)in_sizes; (void)n_in; (void)out_size; (void)ws_size;
  const float* X       = (const float*)d_in[0];
  const float* GUM     = (const float*)d_in[1];
  const float* CEN     = (const float*)d_in[2];
  const float* CM_WIN  = (const float*)d_in[3];
  const float* CM_CW   = (const float*)d_in[4];
  const float* CM_CB   = (const float*)d_in[5];
  const float* CM_WX   = (const float*)d_in[6];
  const float* CM_WDT  = (const float*)d_in[7];
  const float* CM_BDT  = (const float*)d_in[8];
  const float* CM_ALOG = (const float*)d_in[9];
  const float* CM_D    = (const float*)d_in[10];
  const float* CM_WOUT = (const float*)d_in[11];
  const float* GM_WIN  = (const float*)d_in[12];
  const float* GM_CW   = (const float*)d_in[13];
  const float* GM_CB   = (const float*)d_in[14];
  const float* GM_WX   = (const float*)d_in[15];
  const float* GM_WDT  = (const float*)d_in[16];
  const float* GM_BDT  = (const float*)d_in[17];
  const float* GM_ALOG = (const float*)d_in[18];
  const float* GM_D    = (const float*)d_in[19];
  const float* GM_WOUT = (const float*)d_in[20];
  const float* CN_G    = (const float*)d_in[21];
  const float* CN_B    = (const float*)d_in[22];
  const float* GN_G    = (const float*)d_in[23];
  const float* GN_B    = (const float*)d_in[24];
  const float* AW1     = (const float*)d_in[25];
  const float* AB1     = (const float*)d_in[26];
  const float* AW2     = (const float*)d_in[27];
  const float* AB2     = (const float*)d_in[28];
  const float* FW1     = (const float*)d_in[29];
  const float* FB1     = (const float*)d_in[30];
  const float* FW2     = (const float*)d_in[31];
  const float* FB2     = (const float*)d_in[32];

  char* p = (char*)d_ws;
  auto alloc = [&](size_t bytes) -> void* {
    void* r = (void*)p; p += (bytes + 255) & ~(size_t)255; return r;
  };
  unsigned short* winbf  = (unsigned short*)alloc((size_t)512 * 128 * 2);
  unsigned short* wxbf   = (unsigned short*)alloc((size_t)48 * 256 * 2);
  unsigned short* woutbf = (unsigned short*)alloc((size_t)128 * 256 * 2);
  int*            idx    = (int*)alloc((size_t)BATCH * NPIX * 4);
  unsigned short* masked = (unsigned short*)alloc((size_t)ROWS * 128 * 2);
  float*          xz     = (float*)alloc((size_t)ROWS * 512 * 4);
  unsigned short* xibf   = (unsigned short*)alloc((size_t)ROWS * 256 * 2);  // reused for gated y
  float*          dbc    = (float*)alloc((size_t)ROWS * DBC_LD * 4);
  float*          dtys   = (float*)alloc((size_t)ROWS * 256 * 4);           // dt, overwritten by y
  float*          upd    = (float*)alloc((size_t)ROWS * 128 * 4);
  float*          scores = (float*)alloc((size_t)ROWS * 4);
  float*          aw     = (float*)alloc((size_t)ROWS * 4);
  float*          reps   = (float*)alloc((size_t)BATCH * 8 * 128 * 4);
  float*          ctx    = (float*)alloc((size_t)BATCH * 8 * 128 * 4);
  float*          fwt    = (float*)alloc((size_t)BATCH * 8 * 4);
  // chunk-scan summaries: (pair, chunk, d, s)
  const size_t nsum = (size_t)NPAIR * NCHUNK * DIN * DST;
  float*          Eb     = (float*)alloc(nsum * 4);
  float*          H0b    = (float*)alloc(nsum * 4);
  float*          HIN    = (float*)alloc(nsum * 4);

  // weights -> bf16 (wx zero-padded 40->48 rows)
  k_cvt_bf16<<<(512 * 128 + 255) / 256, 256, 0, stream>>>(CM_WIN, winbf, 512 * 128);
  k_pad_wx<<<48, 256, 0, stream>>>(CM_WX, wxbf);
  k_cvt_bf16<<<(128 * 256 + 255) / 256, 256, 0, stream>>>(CM_WOUT, woutbf, 128 * 256);

  // hard gumbel assignment + 8 masked bf16 copies
  k_assign<<<(BATCH * NPIX) / 8, 256, 0, stream>>>(X, GUM, CEN, idx, masked);

  // xz = masked @ win^T          (65536 x 512, K=128)  -- WMMA bf16
  wmma_gemm_bf16<<<(ROWS / 16) * (512 / 16), 32, 0, stream>>>(masked, winbf, xz, ROWS, 512, 128);

  // depthwise conv + silu -> xi (bf16)
  k_conv_silu<<<(ROWS * 256) / 256, 256, 0, stream>>>(xz, CM_CW, CM_CB, xibf);

  // dbc = xi @ wx^T              (65536 x 48, K=256)   -- WMMA bf16
  wmma_gemm_bf16<<<(ROWS / 16) * (DBC_LD / 16), 32, 0, stream>>>(xibf, wxbf, dbc, ROWS, DBC_LD, 256);

  // dt = softplus(dbc[:, :8] @ wdt^T + bdt)
  k_dt<<<(ROWS * 256) / 256, 256, 0, stream>>>(dbc, CM_WDT, CM_BDT, dtys);

  // chunked parallel selective scan: 1024-way parallel passes, 64-deep serial
  k_scan1<<<NPAIR * NCHUNK, 256, 0, stream>>>(dbc, dtys, xibf, CM_ALOG, Eb, H0b);
  k_scan2<<<NPAIR, 256, 0, stream>>>(Eb, H0b, HIN);
  k_scan3<<<NPAIR * NCHUNK, 256, 0, stream>>>(dbc, dtys, xibf, CM_ALOG, CM_D, HIN);

  // gate with silu(z) -> bf16 (reuse xibf)
  k_gate<<<(ROWS * 256) / 256, 256, 0, stream>>>(dtys, xz, xibf);

  // upd = y_gated @ wout^T       (65536 x 128, K=256)  -- WMMA bf16
  wmma_gemm_bf16<<<(ROWS / 16) * (128 / 16), 32, 0, stream>>>(xibf, woutbf, upd, ROWS, 128, 256);

  // cn layernorm (in-place)
  k_layernorm<<<ROWS / 8, 256, 0, stream>>>(upd, CN_G, CN_B);

  // attention scores + mask penalty, softmax over pixels, weighted pool
  k_score<<<ROWS / 8, 256, 0, stream>>>(upd, AW1, AB1, AW2, AB2, idx, scores);
  k_softmax_aw<<<NPAIR, 256, 0, stream>>>(scores, aw);
  k_reps<<<NPAIR, 128, 0, stream>>>(aw, upd, reps);

  // global mamba over 8 reps + gn layernorm + forget-gate MLP
  k_global<<<BATCH, 256, 0, stream>>>(reps, GM_WIN, GM_CW, GM_CB, GM_WX, GM_WDT, GM_BDT,
                                      GM_ALOG, GM_D, GM_WOUT, GN_G, GN_B,
                                      FW1, FB1, FW2, FB2, ctx, fwt);

  // final combine
  k_out<<<(BATCH * NPIX * 128) / 256, 256, 0, stream>>>(upd, idx, fwt, ctx, (float*)d_out);
}
